// CombinedStage2FairnessLoss_63230508532321
// MI455X (gfx1250) — matrix operand order repaired and checked by reading, not compiled, not run on hardware
//
#include <hip/hip_runtime.h>
#include <hip/hip_bf16.h>
#include <math.h>

// ---------------------------------------------------------------------------
// Debiased Sinkhorn fairness loss on MI455X (gfx1250, wave32, WMMA).
// Phase 1: deterministic ordered gather of (class,subgroup) cells.
// Phase 2: pack features to f16, compute per-row squared norms (f32).
// Phase 3: all cost matrices via v_wmma_f32_16x16x32_f16 (one 16x16 tile/wave),
//          writing both C and C^T so every softmin row-read is coalesced.
// Phase 4: 122 eps-scaling iterations, one wave per softmin row, L2-resident C.
// Phase 5: final extrapolation + mean reduction.
// Workspace requirement: ~106 MB (C matrices sized to stay in the 192MB L2).
// ---------------------------------------------------------------------------

typedef _Float16 v16h __attribute__((ext_vector_type(16)));
typedef _Float16 v8h  __attribute__((ext_vector_type(8)));
typedef float    v8f  __attribute__((ext_vector_type(8)));

#define NSG 8
#define DIMF 256
#define ML 256           // local cell size
#define MG 512           // global cell size
#define NPAIR 28
#define NLP 56           // local pairs (2 classes * 28)

__constant__ int c_pi[28] = {0,0,0,0,0,0,0,1,1,1,1,1,1,2,2,2,2,2,3,3,3,3,4,4,4,5,5,6};
__constant__ int c_pj[28] = {1,2,3,4,5,6,7,2,3,4,5,6,7,3,4,5,6,7,4,5,6,7,5,6,7,6,7,7};

// ---------------- Phase 1: ordered index gather (1 wave per cell) ----------
__global__ void gather_idx_kernel(const int* __restrict__ labels,
                                  const int* __restrict__ subg,
                                  int* __restrict__ lidx,   // [16][256]
                                  int* __restrict__ gidx)   // [8][512]
{
    const int cell = blockIdx.x;      // 0..23
    const int lane = threadIdx.x;     // blockDim = 32
    int count = 0;
    if (cell < 16) {
        const int lbl = cell >> 3, sg = cell & 7;
        for (int base = 0; base < 4096; base += 32) {
            const int i = base + lane;
            const bool pred = (labels[i] == lbl) && (subg[i] == sg);
            const unsigned long long mask = __ballot(pred);
            const int prefix = __popcll(mask & ((1ull << lane) - 1ull));
            if (pred && (count + prefix) < ML) lidx[cell * ML + count + prefix] = i;
            count += __popcll(mask);
        }
    } else {
        const int sg = cell - 16;
        for (int base = 0; base < 4096; base += 32) {
            const int i = base + lane;
            const bool pred = (subg[i] == sg);
            const unsigned long long mask = __ballot(pred);
            const int prefix = __popcll(mask & ((1ull << lane) - 1ull));
            if (pred && (count + prefix) < MG) gidx[sg * MG + count + prefix] = i;
            count += __popcll(mask);
        }
    }
}

// ---------------- Phase 2: f16 pack + squared norms ------------------------
__global__ void gather_feat_kernel(const float* __restrict__ feats,
                                   const int* __restrict__ lidx,
                                   const int* __restrict__ gidx,
                                   _Float16* __restrict__ lH,   // [16*256][256]
                                   _Float16* __restrict__ gH,   // [8*512][256]
                                   float* __restrict__ x2l,     // [16*256]
                                   float* __restrict__ x2g)     // [8*512]
{
    const int b = blockIdx.x;         // 0..8191
    const int t = threadIdx.x;        // 256
    int srcRow; _Float16* dst; float* x2dst;
    if (b < 4096) { srcRow = lidx[b]; dst = lH + (size_t)b * DIMF; x2dst = x2l + b; }
    else { const int u = b - 4096; srcRow = gidx[u]; dst = gH + (size_t)u * DIMF; x2dst = x2g + u; }
    const float v = feats[(size_t)srcRow * DIMF + t];
    dst[t] = (_Float16)v;
    float p = v * v;
    for (int off = 16; off; off >>= 1) p += __shfl_xor(p, off, 32);
    __shared__ float ws8[8];
    if ((t & 31) == 0) ws8[t >> 5] = p;
    __syncthreads();
    if (t == 0) { float s = 0.f; for (int w = 0; w < 8; ++w) s += ws8[w]; *x2dst = s; }
}

// ---------------- Phase 3: WMMA cost matrices ------------------------------
struct CostParams {
    const _Float16* lH; const _Float16* gH;
    const float* x2l;   const float* x2g;
    float *lCxy, *lCyx, *lCxx, *gCxy, *gCyx, *gCxx;
};

// wave tasks: [0,14336) lCxy | [14336,18432) lCxx | [18432,47104) gCxy | [47104,55296) gCxx
__global__ void cost_kernel(CostParams P)
{
    const int wid  = blockIdx.x * 8 + (threadIdx.x >> 5);
    const int lane = threadIdx.x & 31;
    const int l = lane & 15, s = lane >> 4;

    const _Float16 *X, *Y; const float *x2X, *x2Y;
    float *Cout, *CTout; int m, ti, tj;

    if (wid < 14336) {                          // local Cxy
        const int p = wid >> 8, tile = wid & 255; ti = tile >> 4; tj = tile & 15; m = ML;
        const int lbl = p / NPAIR, k = p % NPAIR;
        const int cX = lbl * NSG + c_pi[k], cY = lbl * NSG + c_pj[k];
        X = P.lH + (size_t)cX * ML * DIMF;  Y = P.lH + (size_t)cY * ML * DIMF;
        x2X = P.x2l + cX * ML;              x2Y = P.x2l + cY * ML;
        Cout  = P.lCxy + (size_t)p * ML * ML;
        CTout = P.lCyx + (size_t)p * ML * ML;
    } else if (wid < 18432) {                   // local Cxx (symmetric)
        const int u = wid - 14336;
        const int c = u >> 8, tile = u & 255; ti = tile >> 4; tj = tile & 15; m = ML;
        X = Y = P.lH + (size_t)c * ML * DIMF;
        x2X = x2Y = P.x2l + c * ML;
        Cout = P.lCxx + (size_t)c * ML * ML;  CTout = nullptr;
    } else if (wid < 47104) {                   // global Cxy
        const int u = wid - 18432;
        const int p = u >> 10, tile = u & 1023; ti = tile >> 5; tj = tile & 31; m = MG;
        const int cX = c_pi[p], cY = c_pj[p];
        X = P.gH + (size_t)cX * MG * DIMF;  Y = P.gH + (size_t)cY * MG * DIMF;
        x2X = P.x2g + cX * MG;              x2Y = P.x2g + cY * MG;
        Cout  = P.gCxy + (size_t)p * MG * MG;
        CTout = P.gCyx + (size_t)p * MG * MG;
    } else {                                    // global Cxx
        const int u = wid - 47104;
        const int c = u >> 10, tile = u & 1023; ti = tile >> 5; tj = tile & 31; m = MG;
        X = Y = P.gH + (size_t)c * MG * DIMF;
        x2X = x2Y = P.x2g + c * MG;
        Cout = P.gCxx + (size_t)c * MG * MG;  CTout = nullptr;
    }

    const _Float16* Arow = X + (size_t)(ti * 16 + l) * DIMF;
    const _Float16* Brow = Y + (size_t)(tj * 16 + l) * DIMF;

    v8f acc = {};
    #pragma unroll
    for (int kb = 0; kb < DIMF; kb += 32) {
        // A fragment (16x32 f16): lane l+16s holds row l, K = {8s..8s+7, 16+8s..16+8s+7}
        const v8h a0 = *(const v8h*)(Arow + kb + 8 * s);
        const v8h a1 = *(const v8h*)(Arow + kb + 16 + 8 * s);
        // B fragment (32x16 f16): lane l+16s holds col l, K = {8s..8s+7, 16+8s..16+8s+7}
        const v8h b0 = *(const v8h*)(Brow + kb + 8 * s);
        const v8h b1 = *(const v8h*)(Brow + kb + 16 + 8 * s);
        v16h a, b;
        #pragma unroll
        for (int e = 0; e < 8; ++e) { a[e] = a0[e]; a[e + 8] = a1[e]; b[e] = b0[e]; b[e + 8] = b1[e]; }
        acc = __builtin_amdgcn_wmma_f32_16x16x32_f16(false, a, false, b, (short)0, acc, false, false);
    }

    // D layout: lane l+16s, vgpr v -> element (M = v+8s, N = l)
    #pragma unroll
    for (int v = 0; v < 8; ++v) {
        const int i = ti * 16 + v + 8 * s;
        const int j = tj * 16 + l;
        const float d  = x2X[i] + x2Y[j] - 2.0f * acc[v];
        const float cv = 0.5f * fmaxf(d, 0.0f);
        Cout[(size_t)i * m + j] = cv;
        if (CTout) CTout[(size_t)j * m + i] = cv;
    }
}

// ---------------- Phase 4: Sinkhorn iteration (1 wave per softmin row) -----
struct IterParams {
    const float *lCxy, *lCyx, *lCxx, *gCxy, *gCyx, *gCxx;
    const float *lf_s, *lg_s, *lp_s, *gf_s, *gg_s, *gp_s;
    float *lf_d, *lg_d, *lp_d, *gf_d, *gg_d, *gp_d;
};

template <int NV>   // NV float4 chunks per lane: 2 -> m=256, 4 -> m=512
__device__ inline float lse_row(const float* __restrict__ Crow,
                                const float* __restrict__ h,
                                float inv_eps, int lane)
{
    float z[NV * 4];
    const float4* C4 = (const float4*)Crow;
    const float4* H4 = (const float4*)h;
    #pragma unroll
    for (int q = 0; q < NV; ++q) {
        const float4 c = C4[q * 32 + lane];
        const float4 g = H4[q * 32 + lane];
        z[q * 4 + 0] = (g.x - c.x) * inv_eps;
        z[q * 4 + 1] = (g.y - c.y) * inv_eps;
        z[q * 4 + 2] = (g.z - c.z) * inv_eps;
        z[q * 4 + 3] = (g.w - c.w) * inv_eps;
    }
    float mx = -3.402823466e38f;
    #pragma unroll
    for (int k = 0; k < NV * 4; ++k) mx = fmaxf(mx, z[k]);
    for (int off = 16; off; off >>= 1) mx = fmaxf(mx, __shfl_xor(mx, off, 32));
    float ss = 0.f;
    #pragma unroll
    for (int k = 0; k < NV * 4; ++k) ss += __expf(z[k] - mx);
    for (int off = 16; off; off >>= 1) ss += __shfl_xor(ss, off, 32);
    return mx + __logf(ss);
}

// wave tasks (65536 total):
// [0,14336) local f | [14336,28672) local g | [28672,32768) local self
// [32768,47104) global f | [47104,61440) global g | [61440,65536) global self
__global__ void sink_iter_kernel(IterParams P, float eps, int final_mode)
{
    const int wid  = blockIdx.x * 8 + (threadIdx.x >> 5);
    const int lane = threadIdx.x & 31;
    const float inv_eps = 1.0f / eps;

    const float *Crow, *h; float oldv; float* dst; bool big;

    if (wid < 14336) {                       // local f: rows of Cxy, h from g
        const int p = wid >> 8, i = wid & 255;
        Crow = P.lCxy + ((size_t)wid << 8);
        h = P.lg_s + (p << 8); oldv = P.lf_s[wid]; dst = P.lf_d + wid; big = false;
    } else if (wid < 28672) {                // local g: rows of Cyx, h from f
        const int u = wid - 14336, p = u >> 8;
        Crow = P.lCyx + ((size_t)u << 8);
        h = P.lf_s + (p << 8); oldv = P.lg_s[u]; dst = P.lg_d + u; big = false;
    } else if (wid < 32768) {                // local self: rows of Cxx (symmetric)
        const int u = wid - 28672, c = u >> 8;
        Crow = P.lCxx + ((size_t)u << 8);
        h = P.lp_s + (c << 8); oldv = P.lp_s[u]; dst = P.lp_d + u; big = false;
    } else if (wid < 47104) {                // global f
        const int u = wid - 32768, p = u >> 9;
        Crow = P.gCxy + ((size_t)u << 9);
        h = P.gf_s ? (P.gg_s + (p << 9)) : nullptr;
        h = P.gg_s + (p << 9); oldv = P.gf_s[u]; dst = P.gf_d + u; big = true;
    } else if (wid < 61440) {                // global g
        const int u = wid - 47104, p = u >> 9;
        Crow = P.gCyx + ((size_t)u << 9);
        h = P.gf_s + (p << 9); oldv = P.gg_s[u]; dst = P.gg_d + u; big = true;
    } else {                                 // global self
        const int u = wid - 61440, c = u >> 9;
        Crow = P.gCxx + ((size_t)u << 9);
        h = P.gp_s + (c << 9); oldv = P.gp_s[u]; dst = P.gp_d + u; big = true;
    }

    const float loga = big ? -6.2383246250395075f : -5.545177444479562f; // -ln(512), -ln(256)
    float lse;
    if (!big) lse = lse_row<2>(Crow, h, inv_eps, lane);
    else      lse = lse_row<4>(Crow, h, inv_eps, lane);
    const float ft = -eps * (loga + lse);
    const float outv = final_mode ? ft : 0.5f * (oldv + ft);
    if (lane == 0) *dst = outv;
}

// ---------------- Phase 5: reduction ---------------------------------------
__global__ void reduce_loss_kernel(const float* __restrict__ lf, const float* __restrict__ lg,
                                   const float* __restrict__ lp, const float* __restrict__ gf,
                                   const float* __restrict__ gg, const float* __restrict__ gp,
                                   float* __restrict__ out)
{
    const int t = threadIdx.x;   // 512
    float accL = 0.f, accG = 0.f;
    for (int u = t; u < NLP * ML; u += 512) {
        const int p = u >> 8, i = u & 255;
        const int lbl = p / NPAIR, k = p % NPAIR;
        accL += lf[u] - lp[(lbl * NSG + c_pi[k]) * ML + i];
        accL += lg[u] - lp[(lbl * NSG + c_pj[k]) * ML + i];
    }
    for (int u = t; u < NPAIR * MG; u += 512) {
        const int p = u >> 9, i = u & 511;
        accG += gf[u] - gp[c_pi[p] * MG + i];
        accG += gg[u] - gp[c_pj[p] * MG + i];
    }
    for (int off = 16; off; off >>= 1) {
        accL += __shfl_xor(accL, off, 32);
        accG += __shfl_xor(accG, off, 32);
    }
    __shared__ float sL[16], sG[16];
    const int w = t >> 5;
    if ((t & 31) == 0) { sL[w] = accL; sG[w] = accG; }
    __syncthreads();
    if (t == 0) {
        float L = 0.f, G = 0.f;
        for (int i = 0; i < 16; ++i) { L += sL[i]; G += sG[i]; }
        out[0] = 1.0f * (L / (float)(NLP * ML)) + 0.5f * (G / (float)(NPAIR * MG));
    }
}

// ---------------------------------------------------------------------------
extern "C" void kernel_launch(void* const* d_in, const int* in_sizes, int n_in,
                              void* d_out, int out_size, void* d_ws, size_t ws_size,
                              hipStream_t stream)
{
    (void)in_sizes; (void)n_in; (void)out_size;
    const float* feats  = (const float*)d_in[0];
    const int*   labels = (const int*)d_in[1];
    const int*   subg   = (const int*)d_in[2];
    float* out = (float*)d_out;

    // ---- workspace partition (float slots); total ~26.4M floats (~106 MB) ----
    float* ws = (float*)d_ws;
    size_t o = 0;
    auto take = [&](size_t n) { float* p = ws + o; o += n; return p; };
    float* lCxy = take(56ull * 65536);
    float* lCyx = take(56ull * 65536);
    float* lCxx = take(16ull * 65536);
    float* gCxy = take(28ull * 262144);
    float* gCyx = take(28ull * 262144);
    float* gCxx = take(8ull * 262144);
    _Float16* lH = (_Float16*)take(16ull * 256 * 256 / 2);
    _Float16* gH = (_Float16*)take(8ull * 512 * 256 / 2);
    float* x2l = take(4096);
    float* x2g = take(4096);
    float* pots = take(2ull * 65536);   // ping-pong potentials
    float* fins = take(65536);          // final extrapolation outputs
    int* lidx = (int*)take(4096);
    int* gidx = (int*)take(4096);
    (void)ws_size;   // requires >= o * 4 bytes (~106 MB)

    // zero parity-0 potentials (initial f=g=p=0)
    hipMemsetAsync(pots, 0, 65536 * sizeof(float), stream);

    gather_idx_kernel<<<24, 32, 0, stream>>>(labels, subg, lidx, gidx);
    gather_feat_kernel<<<8192, 256, 0, stream>>>(feats, lidx, gidx, lH, gH, x2l, x2g);

    CostParams cp;
    cp.lH = lH; cp.gH = gH; cp.x2l = x2l; cp.x2g = x2g;
    cp.lCxy = lCxy; cp.lCyx = lCyx; cp.lCxx = lCxx;
    cp.gCxy = gCxy; cp.gCyx = gCyx; cp.gCxx = gCxx;
    cost_kernel<<<6912, 256, 0, stream>>>(cp);   // 55296 wave tiles

    // eps schedule: eps0=4*DIM=1024 down to blur^2=1e-8 by 0.81x
    const double eps0 = 1024.0, ratio = 0.81, target = 1e-8;
    const int n = (int)ceil(log(target / eps0) / log(ratio));   // 121 -> 122 values

    IterParams ip;
    ip.lCxy = lCxy; ip.lCyx = lCyx; ip.lCxx = lCxx;
    ip.gCxy = gCxy; ip.gCyx = gCyx; ip.gCxx = gCxx;
    auto setSrc = [&](float* s) {
        ip.lf_s = s; ip.lg_s = s + 14336; ip.lp_s = s + 28672;
        ip.gf_s = s + 32768; ip.gg_s = s + 47104; ip.gp_s = s + 61440;
    };
    auto setDst = [&](float* d) {
        ip.lf_d = d; ip.lg_d = d + 14336; ip.lp_d = d + 28672;
        ip.gf_d = d + 32768; ip.gg_d = d + 47104; ip.gp_d = d + 61440;
    };

    for (int t = 0; t <= n; ++t) {
        const float eps = (float)fmax(eps0 * pow(ratio, (double)t), target);
        setSrc(pots + (size_t)(t & 1) * 65536);
        setDst(pots + (size_t)((t + 1) & 1) * 65536);
        sink_iter_kernel<<<8192, 256, 0, stream>>>(ip, eps, 0);
    }

    // final extrapolation at eps_f = last eps, no averaging
    const float eps_f = (float)fmax(eps0 * pow(ratio, (double)n), target);
    setSrc(pots + (size_t)((n + 1) & 1) * 65536);
    setDst(fins);
    sink_iter_kernel<<<8192, 256, 0, stream>>>(ip, eps_f, 1);

    reduce_loss_kernel<<<1, 512, 0, stream>>>(fins, fins + 14336, fins + 28672,
                                              fins + 32768, fins + 47104, fins + 61440, out);
}
